// MultiHeadAttention_91225105367795
// MI455X (gfx1250) — compile-verified, matrix operand order checked by reading, not run on hardware
//
#include <hip/hip_runtime.h>
#include <hip/hip_bf16.h>

typedef __attribute__((ext_vector_type(16))) _Float16 v16h;
typedef __attribute__((ext_vector_type(8)))  _Float16 v8h;
typedef __attribute__((ext_vector_type(4)))  _Float16 v4h;
typedef __attribute__((ext_vector_type(8)))  float    v8f;
typedef __attribute__((ext_vector_type(4)))  int      v4i;

constexpr int D  = 1024;   // model dim
constexpr int H  = 16;     // heads
constexpr int DK = 64;     // head dim
constexpr int S  = 2048;   // sequence
constexpr int B  = 2;      // batch
constexpr int M  = B * S;  // 4096 rows for projection GEMMs

// ------------------------------------------------ CDNA5 async-LDS helpers ---

#if defined(__has_builtin)
#  if __has_builtin(__builtin_amdgcn_global_load_async_to_lds_b128)
#    define HAVE_ASYNC_LDS 1
#  endif
#  if __has_builtin(__builtin_amdgcn_s_wait_asynccnt)
#    define HAVE_WAIT_ASYNC 1
#  endif
#endif
#ifndef HAVE_ASYNC_LDS
#  define HAVE_ASYNC_LDS 0
#endif
#ifndef HAVE_WAIT_ASYNC
#  define HAVE_WAIT_ASYNC 0
#endif

// 16B per-lane DMA: global -> LDS, tracked by ASYNCcnt.
// Probe-confirmed param types: (v4i addrspace(1)*, v4i addrspace(3)*, Imm, Imm)
__device__ __forceinline__ void async_copy16(const void* g, void* l) {
#if HAVE_ASYNC_LDS
  __builtin_amdgcn_global_load_async_to_lds_b128(
      (__attribute__((address_space(1))) v4i*)g,
      (__attribute__((address_space(3))) v4i*)l, /*offset=*/0, /*cpol=*/0);
#else
  *(uint4*)l = *(const uint4*)g;  // synchronous fallback
#endif
}

__device__ __forceinline__ void wait_async0() {
#if HAVE_WAIT_ASYNC
  __builtin_amdgcn_s_wait_asynccnt(0);
#elif HAVE_ASYNC_LDS
  asm volatile("s_wait_asynccnt 0" ::: "memory");
#endif
}

// ---------------------------------------------------------------- helpers ---

__device__ __forceinline__ v8f zero_v8f() {
  v8f z;
#pragma unroll
  for (int i = 0; i < 8; ++i) z[i] = 0.0f;
  return z;
}

__device__ __forceinline__ v8f wmma16x16x32(v16h a, v16h b, v8f c) {
  // D = A(16x32 f16) * B(32x16 f16) + C(16x16 f32)
  return __builtin_amdgcn_wmma_f32_16x16x32_f16(
      /*neg_a=*/false, a, /*neg_b=*/false, b,
      /*c_mod=*/(short)0, c, /*reuse_a=*/false, /*reuse_b=*/false);
}

// Load a 16x32 f16 WMMA operand fragment.
// Layout per ISA 7.12.2 (16-bit A 16x32): lane = m + 16*((k%16)>=8),
// in-lane element e = (k%8) + 8*(k/16).  The same loader serves the B
// operand when the source matrix is stored with K contiguous per "n" row.
// Requires: stride % 8 == 0, k_base % 8 == 0, p 16-byte aligned.
__device__ __forceinline__ v16h load_frag(const _Float16* p, int stride,
                                          int row_base, int k_base) {
  const int lane = threadIdx.x & 31;
  const int r    = row_base + (lane & 15);
  const int half = lane >> 4;
  const _Float16* rp = p + (size_t)r * stride + k_base + half * 8;
  union { v16h v; v8h h[2]; } u;
  u.h[0] = *(const v8h*)(rp);        // k = 8*half + 0..7
  u.h[1] = *(const v8h*)(rp + 16);   // k = 16 + 8*half + 0..7
  return u.v;
}

// ------------------------------------------------------------ cast kernels ---

__global__ __launch_bounds__(256) void cast_f32_f16_x4(
    const float* __restrict__ src, _Float16* __restrict__ dst, int n4) {
  int i = blockIdx.x * blockDim.x + threadIdx.x;
  if (i < n4) {
    const float4 f = ((const float4*)src)[i];
    v4h o;
    o[0] = (_Float16)f.x; o[1] = (_Float16)f.y;
    o[2] = (_Float16)f.z; o[3] = (_Float16)f.w;
    ((v4h*)dst)[i] = o;
  }
}

// W [K=D][N=D] f32 row-major  ->  Wt [N][K] f16 row-major (transposed)
__global__ __launch_bounds__(256) void transpose_w(
    const float* __restrict__ W, _Float16* __restrict__ Wt) {
  __shared__ float tile[32][33];
  const int bx = blockIdx.x * 32;  // N
  const int by = blockIdx.y * 32;  // K
  const int tx = threadIdx.x, ty = threadIdx.y;  // 32 x 8
#pragma unroll
  for (int r = 0; r < 4; ++r)
    tile[ty + r * 8][tx] = W[(size_t)(by + ty + r * 8) * D + bx + tx];
  __syncthreads();
#pragma unroll
  for (int r = 0; r < 4; ++r)
    Wt[(size_t)(bx + ty + r * 8) * D + by + tx] = (_Float16)tile[tx][ty + r * 8];
}

// ------------------------------------------------------------- WMMA GEMM ----
// C[M x D] = A[M x D] * W + bias, with W given as Wt = W^T f16 [D(n)][D(k)].
// Block tile 128(M) x 64(N), K-step 32, double-buffered async-LDS staging.
// 8 waves in a 4x2 grid, each wave computes a 32x32 region (2x2 WMMA tiles).
// MODE 0: store f16 to [B][H][S][DK]   (Q / K projections)
// MODE 1: store f16 to [B][H][DK][S]   (V projection, transposed for PV WMMA)
// MODE 2: store f32 to [M][D]          (output projection -> d_out)
template <int MODE>
__global__ __launch_bounds__(256) void gemm_wmma(
    const _Float16* __restrict__ A, const _Float16* __restrict__ Bt,
    const float* __restrict__ bias, void* __restrict__ outp) {
  constexpr int BK = 32;
  __shared__ __align__(16) _Float16 sA[2][128 * BK];
  __shared__ __align__(16) _Float16 sB[2][64 * BK];

  const int tid  = threadIdx.x;
  const int lane = tid & 31;
  const int wid  = tid >> 5;
  const int wm   = wid >> 1;  // 0..3
  const int wn   = wid & 1;   // 0..1
  const int m0   = blockIdx.y * 128;
  const int n0   = blockIdx.x * 64;

  // Issue this block's share of the async tile DMA for K-step `kt`.
  auto stage = [&](int buf, int kt) {
#pragma unroll
    for (int t = 0; t < 2; ++t) {
      const int u = tid + t * 256;
      const int row = u >> 2, c8 = (u & 3) * 8;
      async_copy16(A + (size_t)(m0 + row) * D + kt + c8,
                   &sA[buf][row * BK + c8]);
    }
    {
      const int row = tid >> 2, c8 = (tid & 3) * 8;
      async_copy16(Bt + (size_t)(n0 + row) * D + kt + c8,
                   &sB[buf][row * BK + c8]);
    }
  };

  v8f c[2][2];
#pragma unroll
  for (int i = 0; i < 2; ++i)
#pragma unroll
    for (int j = 0; j < 2; ++j) c[i][j] = zero_v8f();

  stage(0, 0);
  wait_async0();
  __syncthreads();

  for (int kt = 0; kt < D; kt += BK) {
    const int buf = (kt / BK) & 1;
    if (kt + BK < D) stage(buf ^ 1, kt + BK);  // DMA next tile during compute

    const v16h a0 = load_frag(sA[buf], BK, wm * 32, 0);
    const v16h a1 = load_frag(sA[buf], BK, wm * 32 + 16, 0);
    const v16h b0 = load_frag(sB[buf], BK, wn * 32, 0);
    const v16h b1 = load_frag(sB[buf], BK, wn * 32 + 16, 0);
    c[0][0] = wmma16x16x32(a0, b0, c[0][0]);
    c[0][1] = wmma16x16x32(a0, b1, c[0][1]);
    c[1][0] = wmma16x16x32(a1, b0, c[1][0]);
    c[1][1] = wmma16x16x32(a1, b1, c[1][1]);

    wait_async0();
    __syncthreads();
  }

  // Epilogue: C/D layout => lane = n + 16*(m>=8), vgpr e => m%8.
  const int hl = lane >> 4, nl = lane & 15;
#pragma unroll
  for (int j = 0; j < 2; ++j) {
    const int col = n0 + wn * 32 + j * 16 + nl;
    const float bc = bias[col];
#pragma unroll
    for (int i = 0; i < 2; ++i) {
#pragma unroll
      for (int e = 0; e < 8; ++e) {
        const int row = m0 + wm * 32 + i * 16 + e + 8 * hl;
        const float val = c[i][j][e] + bc;
        if (MODE == 2) {
          ((float*)outp)[(size_t)row * D + col] = val;
        } else {
          const int bb = row / S, ss = row % S;
          const int hh = col / DK, dd = col % DK;
          _Float16* o = (_Float16*)outp;
          if (MODE == 0)
            o[(((size_t)bb * H + hh) * S + ss) * DK + dd] = (_Float16)val;
          else
            o[(((size_t)bb * H + hh) * DK + dd) * S + ss] = (_Float16)val;
        }
      }
    }
  }
}

// --------------------------------------------------------- flash attention ---
// Grid: (S/128, H, B), 256 threads (8 waves).  Each wave owns 16 query rows.
// K/V tiles for the whole block are DMA'd into LDS (double-buffered async),
// shared by all 8 waves; probabilities round-trip through a wave-private LDS
// tile to convert C-layout -> A-layout for the PV WMMA.
__global__ __launch_bounds__(256) void attention(
    const _Float16* __restrict__ Qp,  // [B][H][S][DK]
    const _Float16* __restrict__ Kp,  // [B][H][S][DK]
    const _Float16* __restrict__ Vt,  // [B][H][DK][S]
    _Float16* __restrict__ ctx) {     // [B][S][H*DK]
  __shared__ __align__(16) _Float16 sK[2][32 * DK];   // [key][d]   4KB x2
  __shared__ __align__(16) _Float16 sV[2][DK * 32];   // [d][key]   4KB x2
  __shared__ __align__(16) _Float16 sP[8 * 16 * 32];  // wave-private P tiles

  const int tid  = threadIdx.x;
  const int lane = tid & 31;
  const int wid  = tid >> 5;
  const int hl   = lane >> 4;   // row-group half
  const int nl   = lane & 15;   // column within tile
  const int h    = blockIdx.y;
  const int bz   = blockIdx.z;
  const int bh   = bz * H + h;

  const _Float16* Qb = Qp + (size_t)bh * S * DK;
  const _Float16* Kb = Kp + (size_t)bh * S * DK;
  const _Float16* Vb = Vt + (size_t)bh * DK * S;
  const int q0 = blockIdx.x * 128 + wid * 16;

  // Async-stage one 32-key K tile ([32][64]) and V tile ([64][32]).
  auto stage = [&](int buf, int kt) {
    {
      const int row = tid >> 3, c8 = (tid & 7) * 8;  // 32 rows x 64 halves
      async_copy16(Kb + (size_t)(kt + row) * DK + c8,
                   &sK[buf][row * DK + c8]);
    }
    {
      const int row = tid >> 2, c8 = (tid & 3) * 8;  // 64 rows x 32 halves
      async_copy16(Vb + (size_t)row * S + kt + c8,
                   &sV[buf][row * 32 + c8]);
    }
  };

  // Preload this wave's Q operand (16 rows x 64 dims) as two A fragments.
  const v16h aq0 = load_frag(Qb, DK, q0, 0);
  const v16h aq1 = load_frag(Qb, DK, q0, 32);

  v8f acc[4];
#pragma unroll
  for (int t = 0; t < 4; ++t) acc[t] = zero_v8f();
  float rmax[8], rsum[8];
#pragma unroll
  for (int e = 0; e < 8; ++e) { rmax[e] = -3.0e38f; rsum[e] = 0.0f; }

  _Float16* sPw = sP + wid * (16 * 32);
  constexpr float scale = 0.125f;  // 1/sqrt(64)

  stage(0, 0);
  wait_async0();
  __syncthreads();

  for (int kt = 0; kt < S; kt += 32) {
    const int buf = (kt >> 5) & 1;
    if (kt + 32 < S) stage(buf ^ 1, kt + 32);  // DMA next tile during compute

    // ---- scores: two 16x16 tiles over this 32-key block -------------------
    // Load all four K fragments into distinct registers first so the ds
    // loads issue as one clause and overlap the WMMA chain.
    const v16h bk00 = load_frag(sK[buf], DK, 0, 0);
    const v16h bk01 = load_frag(sK[buf], DK, 0, 32);
    const v16h bk10 = load_frag(sK[buf], DK, 16, 0);
    const v16h bk11 = load_frag(sK[buf], DK, 16, 32);
    v8f s0 = zero_v8f(), s1 = zero_v8f();
    s0 = wmma16x16x32(aq0, bk00, s0);
    s0 = wmma16x16x32(aq1, bk01, s0);
    s1 = wmma16x16x32(aq0, bk10, s1);
    s1 = wmma16x16x32(aq1, bk11, s1);

    // ---- online softmax ---------------------------------------------------
    float tmax[8];
#pragma unroll
    for (int e = 0; e < 8; ++e) {
      s0[e] *= scale; s1[e] *= scale;
      tmax[e] = fmaxf(s0[e], s1[e]);
    }
#pragma unroll
    for (int m = 1; m < 16; m <<= 1)
#pragma unroll
      for (int e = 0; e < 8; ++e)
        tmax[e] = fmaxf(tmax[e], __shfl_xor(tmax[e], m, 16));

    float alpha[8];
#pragma unroll
    for (int e = 0; e < 8; ++e) {
      const float mn = fmaxf(rmax[e], tmax[e]);
      alpha[e] = __expf(rmax[e] - mn);
      rmax[e] = mn;
    }

    float p0[8], p1[8], psum[8];
#pragma unroll
    for (int e = 0; e < 8; ++e) {
      p0[e] = __expf(s0[e] - rmax[e]);
      p1[e] = __expf(s1[e] - rmax[e]);
      psum[e] = p0[e] + p1[e];
    }
#pragma unroll
    for (int m = 1; m < 16; m <<= 1)
#pragma unroll
      for (int e = 0; e < 8; ++e)
        psum[e] += __shfl_xor(psum[e], m, 16);
#pragma unroll
    for (int e = 0; e < 8; ++e) rsum[e] = rsum[e] * alpha[e] + psum[e];
#pragma unroll
    for (int t = 0; t < 4; ++t)
#pragma unroll
      for (int e = 0; e < 8; ++e) acc[t][e] *= alpha[e];

    // ---- P: C-layout -> A-layout via wave-private LDS tile ----------------
    // DS ops from one wave execute in order; wave_barrier pins compiler order.
#pragma unroll
    for (int e = 0; e < 8; ++e) {
      sPw[(e + 8 * hl) * 32 + nl]      = (_Float16)p0[e];
      sPw[(e + 8 * hl) * 32 + 16 + nl] = (_Float16)p1[e];
    }
    __builtin_amdgcn_wave_barrier();
    const v16h ap = load_frag(sPw, 32, 0, 0);
    __builtin_amdgcn_wave_barrier();

    // ---- ctx += P(16x32) x V(32x64) from the shared LDS V tile ------------
    const v16h bv0 = load_frag(sV[buf], 32, 0, 0);
    const v16h bv1 = load_frag(sV[buf], 32, 16, 0);
    const v16h bv2 = load_frag(sV[buf], 32, 32, 0);
    const v16h bv3 = load_frag(sV[buf], 32, 48, 0);
    acc[0] = wmma16x16x32(ap, bv0, acc[0]);
    acc[1] = wmma16x16x32(ap, bv1, acc[1]);
    acc[2] = wmma16x16x32(ap, bv2, acc[2]);
    acc[3] = wmma16x16x32(ap, bv3, acc[3]);

    wait_async0();
    __syncthreads();
  }

  // ---- normalize + store ctx [B][S][H*DK] f16 -----------------------------
#pragma unroll
  for (int dt = 0; dt < 4; ++dt)
#pragma unroll
    for (int e = 0; e < 8; ++e) {
      const int srow = q0 + e + 8 * hl;
      const int col  = h * DK + dt * 16 + nl;
      ctx[((size_t)bz * S + srow) * D + col] = (_Float16)(acc[dt][e] / rsum[e]);
    }
}

// ---------------------------------------------------------------- launcher ---

extern "C" void kernel_launch(void* const* d_in, const int* in_sizes, int n_in,
                              void* d_out, int out_size, void* d_ws,
                              size_t ws_size, hipStream_t stream) {
  (void)in_sizes; (void)n_in; (void)out_size; (void)ws_size;

  const float* q  = (const float*)d_in[0];
  const float* k  = (const float*)d_in[1];
  const float* v  = (const float*)d_in[2];
  const float* Wq = (const float*)d_in[3];
  const float* bq = (const float*)d_in[4];
  const float* Wk = (const float*)d_in[5];
  const float* bk = (const float*)d_in[6];
  const float* Wv = (const float*)d_in[7];
  const float* bv = (const float*)d_in[8];
  const float* Wo = (const float*)d_in[9];
  const float* bo = (const float*)d_in[10];

  // Workspace carve-up (all f16, all sizes big powers of two => aligned).
  char* w = (char*)d_ws;
  auto take = [&](size_t halves) {
    _Float16* p = (_Float16*)w;
    w += halves * sizeof(_Float16);
    return p;
  };
  _Float16* q16 = take((size_t)M * D);
  _Float16* k16 = take((size_t)M * D);
  _Float16* v16 = take((size_t)M * D);
  _Float16* Wqt = take((size_t)D * D);
  _Float16* Wkt = take((size_t)D * D);
  _Float16* Wvt = take((size_t)D * D);
  _Float16* Wot = take((size_t)D * D);
  _Float16* Qp  = take((size_t)M * D);   // [B][H][S][DK]
  _Float16* Kp  = take((size_t)M * D);   // [B][H][S][DK]
  _Float16* Vtr = take((size_t)M * D);   // [B][H][DK][S]
  _Float16* ctx = take((size_t)M * D);   // [B][S][D]

  // 1) f32 -> f16 casts of activations.
  const int n4 = (M * D) / 4;
  const int cblocks = n4 / 256;
  cast_f32_f16_x4<<<cblocks, 256, 0, stream>>>(q, q16, n4);
  cast_f32_f16_x4<<<cblocks, 256, 0, stream>>>(k, k16, n4);
  cast_f32_f16_x4<<<cblocks, 256, 0, stream>>>(v, v16, n4);

  // 2) Weight transposes (W -> W^T f16) for K-contiguous B operands.
  dim3 tgrid(D / 32, D / 32), tblk(32, 8);
  transpose_w<<<tgrid, tblk, 0, stream>>>(Wq, Wqt);
  transpose_w<<<tgrid, tblk, 0, stream>>>(Wk, Wkt);
  transpose_w<<<tgrid, tblk, 0, stream>>>(Wv, Wvt);
  transpose_w<<<tgrid, tblk, 0, stream>>>(Wo, Wot);

  // 3) Projections.
  dim3 ggrid(D / 64, M / 128);
  gemm_wmma<0><<<ggrid, 256, 0, stream>>>(q16, Wqt, bq, (void*)Qp);
  gemm_wmma<0><<<ggrid, 256, 0, stream>>>(k16, Wkt, bk, (void*)Kp);
  gemm_wmma<1><<<ggrid, 256, 0, stream>>>(v16, Wvt, bv, (void*)Vtr);

  // 4) Attention.
  attention<<<dim3(S / 128, H, B), 256, 0, stream>>>(Qp, Kp, Vtr, ctx);

  // 5) Output projection straight into d_out (f32).
  gemm_wmma<2><<<ggrid, 256, 0, stream>>>(ctx, Wot, bo, d_out);
}
